// Token2TokenModule_75368086110619
// MI455X (gfx1250) — matrix operand order repaired and checked by reading, not compiled, not run on hardware
//
#include <hip/hip_runtime.h>
#include <hip/hip_bf16.h>
#include <math.h>

typedef __attribute__((ext_vector_type(16))) _Float16 v16h;
typedef __attribute__((ext_vector_type(8)))  _Float16 v8h;
typedef __attribute__((ext_vector_type(8)))  float    v8f;

// ---------------------------------------------------------------------------
// GEMM: C(M,N) = A(M,Kpad f16, row-major) * Bt(N,Kpad f16, transposed)
//       [+ bias] [+ resid].  Requires M%64==0, N%32==0, Kpad%32==0,
//       256B-aligned buffers, lda/ldb multiples of 8 halves.
// One wave per 64x32 C tile: 4 M-subtiles x 2 N-subtiles, 8 WMMAs per K-step.
// Fragments loaded directly from global as contiguous b128 chunks in a
// contiguous k order shared by A(16x32)/B(32x16) (A: row l15, K=16g..+15
// ; B(from Bt): col l15, K=16g..+15).
// ---------------------------------------------------------------------------
__global__ void __launch_bounds__(32)
k_gemm_wmma(const _Float16* __restrict__ A, int lda,
            const _Float16* __restrict__ Bt, int ldb,
            const float* __restrict__ bias,
            const float* __restrict__ resid, int resid_ld, int resid_off,
            float* __restrict__ C, int ldc, int Kpad) {
  const int lane = threadIdx.x;       // 0..31 (wave32)
  const int g    = lane >> 4;
  const int l15  = lane & 15;
  const int mbase = blockIdx.y * 64;
  const int nbase = blockIdx.x * 32;

  const _Float16* Ap = A  + (size_t)(mbase + l15) * lda + 16 * g;
  const _Float16* Bp = Bt + (size_t)(nbase + l15) * ldb + 16 * g;

  v8f acc[4][2] = {};
  for (int k0 = 0; k0 < Kpad; k0 += 32) {
    if (k0 + 32 < Kpad) {
      __builtin_prefetch(Ap + k0 + 32, 0, 1);
      __builtin_prefetch(Bp + k0 + 32, 0, 1);
    }
    v16h a[4], b[2];
#pragma unroll
    for (int mt = 0; mt < 4; ++mt) {
      const _Float16* p = Ap + (size_t)(16 * mt) * lda + k0;
      v8h lo = *(const v8h*)(p);        // K = k0+16g .. +7
      v8h hi = *(const v8h*)(p + 8);    // K = k0+16g+8 .. +15
      a[mt] = __builtin_shufflevector(lo, hi, 0,1,2,3,4,5,6,7,8,9,10,11,12,13,14,15);
    }
#pragma unroll
    for (int nt = 0; nt < 2; ++nt) {
      // contiguous 16 halves: K = k0+16g .. +15 (32B aligned)
      b[nt] = *(const v16h*)(Bp + (size_t)(16 * nt) * ldb + k0);
    }
#pragma unroll
    for (int mt = 0; mt < 4; ++mt)
#pragma unroll
      for (int nt = 0; nt < 2; ++nt)
        acc[mt][nt] = __builtin_amdgcn_wmma_f32_16x16x32_f16(
            false, a[mt], false, b[nt], (short)0, acc[mt][nt], false, false);
  }
  // C layout: VGPR r -> M = r + 8*g ; N = l15
#pragma unroll
  for (int nt = 0; nt < 2; ++nt) {
    int n = nbase + nt * 16 + l15;
    float bv = bias ? bias[n] : 0.0f;
#pragma unroll
    for (int mt = 0; mt < 4; ++mt) {
#pragma unroll
      for (int r = 0; r < 8; ++r) {
        int m = mbase + mt * 16 + r + 8 * g;
        float v = acc[mt][nt][r] + bv;
        if (resid) v += resid[(size_t)m * resid_ld + resid_off + n];
        C[(size_t)m * ldc + n] = v;
      }
    }
  }
}

// f32 weight (K x N) -> f16 TRANSPOSED (N x Kpad), zero pad cols K..Kpad-1
__global__ void k_cvt_w(const float* __restrict__ src, _Float16* __restrict__ dst,
                        int K, int N, int Kpad) {
  int i = blockIdx.x * blockDim.x + threadIdx.x;
  if (i >= Kpad * N) return;
  int n = i / Kpad, k = i % Kpad;
  dst[i] = (_Float16)((k < K) ? src[k * N + n] : 0.0f);
}

// unfold(7,4,2) over (64,3,224,224) fused with layernorm(147) -> f16 rows of 160
__global__ void k_unfold7_ln(const float* __restrict__ x,
                             const float* __restrict__ g1, const float* __restrict__ b1,
                             _Float16* __restrict__ outA) {
  const int l = blockIdx.x;           // 0..3135
  const int b = blockIdx.y;           // 0..63
  const int oh = l / 56, ow = l % 56;
  const int f = threadIdx.x;          // 0..255
  __shared__ float red[256], red2[256];
  float v = 0.0f;
  if (f < 147) {
    int c = f / 49, rem = f % 49, i = rem / 7, j = rem % 7;
    int ih = oh * 4 + i - 2, iw = ow * 4 + j - 2;
    if (ih >= 0 && ih < 224 && iw >= 0 && iw < 224)
      v = x[(((size_t)b * 3 + c) * 224 + ih) * 224 + iw];
  }
  red[f]  = (f < 147) ? v : 0.0f;
  red2[f] = (f < 147) ? v * v : 0.0f;
  __syncthreads();
  for (int s = 128; s > 0; s >>= 1) {
    if (f < s) { red[f] += red[f + s]; red2[f] += red2[f + s]; }
    __syncthreads();
  }
  float mu  = red[0] / 147.0f;
  float var = red2[0] / 147.0f - mu * mu;
  float rs  = rsqrtf(var + 1e-5f);
  size_t row = ((size_t)b * 3136 + l) * 160;
  if (f < 147)       outA[row + f] = (_Float16)((v - mu) * rs * g1[f] + b1[f]);
  else if (f < 160)  outA[row + f] = (_Float16)0.0f;   // K padding
}

// unfold(3,2,1) over (B, Hd*Wd, 64) viewed channel-major, optional layernorm(576)
__global__ void k_unfold3_ln(const float* __restrict__ t,
                             const float* __restrict__ g1, const float* __restrict__ b1,
                             _Float16* __restrict__ outA,
                             int Hd, int Wd, int OW, int Ltot) {
  const int l = blockIdx.x, b = blockIdx.y;
  const int oh = l / OW, ow = l % OW;
  const int tid = threadIdx.x;        // 0..191, 3 features each
  __shared__ float red[192], red2[192];
  const int T = Hd * Wd;
  float vals[3]; float s = 0.f, s2 = 0.f;
#pragma unroll
  for (int q = 0; q < 3; ++q) {
    int f = tid + q * 192;
    int c = f / 9, rem = f % 9, i = rem / 3, j = rem % 3;
    int ih = oh * 2 + i - 1, iw = ow * 2 + j - 1;
    float v = 0.f;
    if (ih >= 0 && ih < Hd && iw >= 0 && iw < Wd)
      v = t[((size_t)b * T + (ih * Wd + iw)) * 64 + c];
    vals[q] = v; s += v; s2 += v * v;
  }
  size_t row = ((size_t)b * Ltot + l) * 576;
  if (g1 == nullptr) {                // final stage: no LN before projection
#pragma unroll
    for (int q = 0; q < 3; ++q) outA[row + tid + q * 192] = (_Float16)vals[q];
    return;
  }
  red[tid] = s; red2[tid] = s2;
  __syncthreads();
  for (int st = 128; st > 0; st >>= 1) {
    if (tid < st && tid + st < 192) { red[tid] += red[tid + st]; red2[tid] += red2[tid + st]; }
    __syncthreads();
  }
  float mu  = red[0] / 576.0f;
  float var = red2[0] / 576.0f - mu * mu;
  float rs  = rsqrtf(var + 1e-5f);
#pragma unroll
  for (int q = 0; q < 3; ++q) {
    int f = tid + q * 192;
    outA[row + f] = (_Float16)((vals[q] - mu) * rs * g1[f] + b1[f]);
  }
}

// rowwise layernorm over 64, f32 in -> f16 out
__global__ void k_ln64(const float* __restrict__ X, const float* __restrict__ g,
                       const float* __restrict__ bt, _Float16* __restrict__ Y) {
  int row = blockIdx.x, f = threadIdx.x;    // block 64
  __shared__ float red[64], red2[64];
  float v = X[(size_t)row * 64 + f];
  red[f] = v; red2[f] = v * v;
  __syncthreads();
  for (int s = 32; s > 0; s >>= 1) {
    if (f < s) { red[f] += red[f + s]; red2[f] += red2[f + s]; }
    __syncthreads();
  }
  float mu = red[0] / 64.f, var = red2[0] / 64.f - mu * mu;
  float rs = rsqrtf(var + 1e-5f);
  Y[(size_t)row * 64 + f] = (_Float16)((v - mu) * rs * g[f] + bt[f]);
}

// kp/qp = exp(w@x - 0.5|x|^2)/sqrt(32); also ksum[b,m] += kp (atomic)
__global__ void k_prm_exp(const float* __restrict__ qkv, const float* __restrict__ w,
                          float* __restrict__ kp, float* __restrict__ qp,
                          float* __restrict__ ksum, int T) {
  int b = blockIdx.y;
  size_t tok = (size_t)b * T + blockIdx.x;
  int m = threadIdx.x;                // 0..31
  __shared__ float kv[64], qv[64], red[32];
  kv[m] = qkv[tok * 192 + m];        kv[m + 32] = qkv[tok * 192 + 32 + m];
  qv[m] = qkv[tok * 192 + 64 + m];   qv[m + 32] = qkv[tok * 192 + 96 + m];
  __syncthreads();
  red[m] = kv[m] * kv[m] + kv[m + 32] * kv[m + 32];
  __syncthreads();
  for (int s = 16; s > 0; s >>= 1) { if (m < s) red[m] += red[m + s]; __syncthreads(); }
  float xdk = 0.5f * red[0];
  __syncthreads();
  red[m] = qv[m] * qv[m] + qv[m + 32] * qv[m + 32];
  __syncthreads();
  for (int s = 16; s > 0; s >>= 1) { if (m < s) red[m] += red[m + s]; __syncthreads(); }
  float xdq = 0.5f * red[0];
  float dk = 0.f, dq = 0.f;
#pragma unroll
  for (int d = 0; d < 64; ++d) {
    float wv = w[m * 64 + d];
    dk += kv[d] * wv; dq += qv[d] * wv;
  }
  const float sc = 0.17677669529663687f;  // 1/sqrt(32)
  float kpe = expf(dk - xdk) * sc;
  float qpe = expf(dq - xdq) * sc;
  kp[tok * 32 + m] = kpe;
  qp[tok * 32 + m] = qpe;
  atomicAdd(&ksum[b * 32 + m], kpe);
}

// kptv[b,d,m] = sum_t v[b,t,d]*kp[b,t,m]   (v lives at qkv[...,128+d])
__global__ void k_kptv(const float* __restrict__ qkv, const float* __restrict__ kp,
                       float* __restrict__ kptv, int T) {
  int b = blockIdx.y;
  int p = blockIdx.x * 256 + threadIdx.x;   // 0..2047
  int d = p >> 5, m = p & 31;
  float acc = 0.f;
  size_t base = (size_t)b * T;
  for (int t = 0; t < T; ++t)
    acc += qkv[(base + t) * 192 + 128 + d] * kp[(base + t) * 32 + m];
  kptv[((size_t)b * 64 + d) * 32 + m] = acc;
}

// y[tok,d] = (qp . kptv[d,:]) / (qp . ksum + 1e-8)  -> f16
__global__ void k_y(const float* __restrict__ qp, const float* __restrict__ ksum,
                    const float* __restrict__ kptv, _Float16* __restrict__ y16, int T) {
  int b = blockIdx.y;
  size_t tok = (size_t)b * T + blockIdx.x;
  int d = threadIdx.x;                // block 64
  __shared__ float qs[32], dr[32];
  if (d < 32) { qs[d] = qp[tok * 32 + d]; dr[d] = qs[d] * ksum[b * 32 + d]; }
  __syncthreads();
  for (int s = 16; s > 0; s >>= 1) { if (d < s) dr[d] += dr[d + s]; __syncthreads(); }
  float D = dr[0] + 1e-8f;
  float acc = 0.f;
#pragma unroll
  for (int m = 0; m < 32; ++m) acc += qs[m] * kptv[((size_t)b * 64 + d) * 32 + m];
  y16[tok * 64 + d] = (_Float16)(acc / D);
}

// exact GELU, f32 -> f16
__global__ void k_gelu(const float* __restrict__ X, _Float16* __restrict__ Y, size_t n) {
  size_t i = (size_t)blockIdx.x * blockDim.x + threadIdx.x;
  if (i < n) {
    float v = X[i];
    Y[i] = (_Float16)(0.5f * v * (1.0f + erff(v * 0.70710678118654752f)));
  }
}

static inline void launch_gemm(const _Float16* A, int lda, const _Float16* Bt, int ldb,
                               const float* bias, const float* resid, int rld, int roff,
                               float* C, int ldc, int M, int Kpad, int N, hipStream_t s) {
  dim3 grid(N / 32, M / 64);
  k_gemm_wmma<<<grid, 32, 0, s>>>(A, lda, Bt, ldb, bias, resid, rld, roff, C, ldc, Kpad);
}

extern "C" void kernel_launch(void* const* d_in, const int* in_sizes, int n_in,
                              void* d_out, int out_size, void* d_ws, size_t ws_size,
                              hipStream_t stream) {
  (void)in_sizes; (void)n_in; (void)out_size; (void)ws_size;
  const float* x = (const float*)d_in[0];
  const float* p1[12]; for (int i = 0; i < 12; ++i) p1[i] = (const float*)d_in[1 + i];
  const float* w1 = (const float*)d_in[13];
  const float* p2[12]; for (int i = 0; i < 12; ++i) p2[i] = (const float*)d_in[14 + i];
  const float* w2 = (const float*)d_in[26];
  const float* pw = (const float*)d_in[27];
  const float* pb = (const float*)d_in[28];

  const int B = 64;
  const size_t TOK1 = (size_t)B * 3136;

  char* ws = (char*)d_ws;
  auto alloc = [&](size_t bytes) -> char* {
    char* p = ws; ws += (bytes + 255) & ~((size_t)255); return p;
  };
  _Float16* bufA    = (_Float16*)alloc(TOK1 * 160 * sizeof(_Float16)); // >= TOK2*576
  float*    bufQKV  = (float*)   alloc(TOK1 * 192 * sizeof(float));    // qkv / FFN scratch
  float*    bufKP   = (float*)   alloc(TOK1 * 32  * sizeof(float));
  float*    bufQP   = (float*)   alloc(TOK1 * 32  * sizeof(float));
  float*    bufKSUM = (float*)   alloc(64 * 32    * sizeof(float));
  float*    bufKPTV = (float*)   alloc(64 * 64 * 32 * sizeof(float));
  _Float16* bufY16  = (_Float16*)alloc(TOK1 * 64  * sizeof(_Float16));
  _Float16* bufH16  = (_Float16*)alloc(TOK1 * 64  * sizeof(_Float16));
  float*    bufT    = (float*)   alloc(TOK1 * 64  * sizeof(float));    // residual stream
  _Float16* bufW    = (_Float16*)alloc(576 * 384  * sizeof(_Float16)); // big weight (f16, transposed)
  _Float16* bufWp   = (_Float16*)alloc(64 * 64    * sizeof(_Float16));
  _Float16* bufW1f  = (_Float16*)alloc(64 * 64    * sizeof(_Float16));
  _Float16* bufW2f  = (_Float16*)alloc(64 * 64    * sizeof(_Float16));

  auto stage = [&](const float* const* p, const float* w, int T, int cin, int Kpad) {
    const size_t TOK = (size_t)B * T;
    // qkv = ln(x) @ qkv_w + qkv_b   (A = bufA, prepared by unfold+ln)
    k_cvt_w<<<dim3((Kpad * 192 + 255) / 256), 256, 0, stream>>>(p[2], bufW, cin, 192, Kpad);
    launch_gemm(bufA, Kpad, bufW, Kpad, p[3], nullptr, 0, 0, bufQKV, 192, (int)TOK, Kpad, 192, stream);
    // performer features
    hipMemsetAsync(bufKSUM, 0, 64 * 32 * sizeof(float), stream);
    k_prm_exp<<<dim3(T, B), 32, 0, stream>>>(bufQKV, w, bufKP, bufQP, bufKSUM, T);
    k_kptv   <<<dim3(8, B), 256, 0, stream>>>(bufQKV, bufKP, bufKPTV, T);
    k_y      <<<dim3(T, B), 64, 0, stream>>>(bufQP, bufKSUM, bufKPTV, bufY16, T);
    // y2 = v + y @ proj_w + proj_b   (residual v strided inside qkv buffer)
    k_cvt_w<<<dim3((64 * 64 + 255) / 256), 256, 0, stream>>>(p[4], bufWp, 64, 64, 64);
    launch_gemm(bufY16, 64, bufWp, 64, p[5], bufQKV, 192, 128, bufT, 64, (int)TOK, 64, 64, stream);
    // FFN: t = y2 + f2(gelu(f1(ln(y2))))
    k_ln64<<<dim3((int)TOK), 64, 0, stream>>>(bufT, p[6], p[7], bufH16);
    k_cvt_w<<<dim3((64 * 64 + 255) / 256), 256, 0, stream>>>(p[8], bufW1f, 64, 64, 64);
    launch_gemm(bufH16, 64, bufW1f, 64, p[9], nullptr, 0, 0, bufQKV, 64, (int)TOK, 64, 64, stream);
    k_gelu<<<dim3((int)((TOK * 64 + 255) / 256)), 256, 0, stream>>>(bufQKV, bufY16, TOK * 64);
    k_cvt_w<<<dim3((64 * 64 + 255) / 256), 256, 0, stream>>>(p[10], bufW2f, 64, 64, 64);
    launch_gemm(bufY16, 64, bufW2f, 64, p[11], bufT, 64, 0, bufT, 64, (int)TOK, 64, 64, stream);
  };

  // stage 1: unfold(7,4,2)+ln -> performer (cin=147, padded K=160)
  k_unfold7_ln<<<dim3(3136, B), 256, 0, stream>>>(x, p1[0], p1[1], bufA);
  stage(p1, w1, 3136, 147, 160);
  // stage 2: unfold(3,2,1) on 56x56 + ln -> performer (cin=576)
  k_unfold3_ln<<<dim3(28 * 28, B), 192, 0, stream>>>(bufT, p2[0], p2[1], bufA, 56, 56, 28, 28 * 28);
  stage(p2, w2, 784, 576, 576);
  // final: unfold(3,2,1) on 28x28 (no ln) -> proj to 384
  k_unfold3_ln<<<dim3(14 * 14, B), 192, 0, stream>>>(bufT, nullptr, nullptr, bufA, 28, 28, 14, 14 * 14);
  k_cvt_w<<<dim3((576 * 384 + 255) / 256), 256, 0, stream>>>(pw, bufW, 576, 384, 576);
  launch_gemm(bufA, 576, bufW, 576, pb, nullptr, 0, 0, (float*)d_out, 384, B * 196, 576, 384, stream);
}
